// PropSTGformer_71055938945743
// MI455X (gfx1250) — matrix operand order repaired and checked by reading, not compiled, not run on hardware
//
#include <hip/hip_runtime.h>
#include <hip/hip_bf16.h>
#include <stdint.h>

typedef __attribute__((ext_vector_type(16))) __bf16 v16bf;
typedef __attribute__((ext_vector_type(8)))  float  v8f;

#define TOKS 196608   // B*T*N
#define DIM  256
#define E3   768
#define FF   512
#define HD   32
#define NS   2048
#define TT   24

__device__ __forceinline__ uint16_t f2bf(float f) {
  union { float f; uint32_t u; } c; c.f = f;
  uint32_t u = c.u;
  return (uint16_t)((u + 0x7FFFu + ((u >> 16) & 1u)) >> 16);
}
__device__ __forceinline__ float bf2f(uint16_t h) {
  union { uint32_t u; float f; } c; c.u = ((uint32_t)h) << 16;
  return c.f;
}

union Frag { v16bf v; uint4 q[2]; uint16_t h[16]; };

// ---------------- K0a: transpose weights to bf16 ----------------
__global__ void k_prep_weights(const float* __restrict__ wqkv,
                               const float* __restrict__ wout,
                               uint16_t* __restrict__ wqkvT,
                               uint16_t* __restrict__ woutT) {
  int idx = blockIdx.x * 256 + threadIdx.x;
  if (idx < DIM * E3) {                 // wqkvT[d][e] = wqkv[e][d]
    int d = idx / E3, e = idx % E3;
    wqkvT[idx] = f2bf(wqkv[e * DIM + d]);
  }
  int i2 = idx - DIM * E3;
  if (i2 >= 0 && i2 < FF * DIM) {       // woutT[f][d] = wout[d][f]
    int f = i2 / DIM, d = i2 % DIM;
    woutT[i2] = f2bf(wout[d * FF + f]);
  }
}

// ---------------- K0b: x -> bf16 ----------------
__global__ void k_x_to_bf(const float* __restrict__ x, uint16_t* __restrict__ xb) {
  int i = (blockIdx.x * 256 + threadIdx.x) * 4;
  float4 v = *(const float4*)(x + i);
  xb[i] = f2bf(v.x); xb[i + 1] = f2bf(v.y);
  xb[i + 2] = f2bf(v.z); xb[i + 3] = f2bf(v.w);
}

// ---------------- K1: QKV GEMM (196608x256 @ 256x768) ----------------
__global__ void k_qkv_gemm(const uint16_t* __restrict__ xb,
                           const uint16_t* __restrict__ wT,
                           uint16_t* __restrict__ qb, uint16_t* __restrict__ kb,
                           float* __restrict__ vf) {
  __shared__ uint16_t ldsB[8 * 512];          // [kstep][lane*16+e] frag-order
  const int tid = threadIdx.x;
  const int n0 = blockIdx.y * 16;
  const int m0 = blockIdx.x * 128;
  for (int i = 0; i < 16; ++i) {              // stage 256x16 B panel, swizzled
    int idx = tid + i * 256;
    int k = idx >> 4, col = idx & 15, kk = k & 31;
    ldsB[(k >> 5) * 512 + (((kk & 8) ? 16 : 0) + col) * 16 + (((kk & 16) >> 1) | (kk & 7))] =
        wT[k * E3 + n0 + col];
  }
  __syncthreads();
  const int w = tid >> 5, lane = tid & 31, khalf = lane >> 4;
  const uint16_t* ap = xb + (size_t)(m0 + w * 16 + (lane & 15)) * DIM + khalf * 8;
  v8f c = {};
  for (int ks = 0; ks < 8; ++ks) {
    Frag a, b;
    a.q[0] = *(const uint4*)(ap + ks * 32);
    a.q[1] = *(const uint4*)(ap + ks * 32 + 16);
    b.q[0] = *(const uint4*)&ldsB[ks * 512 + lane * 16];
    b.q[1] = *(const uint4*)&ldsB[ks * 512 + lane * 16 + 8];
    c = __builtin_amdgcn_wmma_f32_16x16x32_bf16(false, a.v, false, b.v,
                                                (short)0, c, false, false);
  }
  const int colg = n0 + (lane & 15);
  for (int r = 0; r < 8; ++r) {
    int row = m0 + w * 16 + khalf * 8 + r;
    float val = c[r];
    if (colg < DIM)          qb[(size_t)row * DIM + colg] = f2bf(val);
    else if (colg < 2 * DIM) kb[(size_t)row * DIM + colg - DIM] = f2bf(val);
    else                     vf[(size_t)row * DIM + colg - 2 * DIM] = val;
  }
}

// ---------------- K2: per-(token,head) L2 normalize q,k (in place) ----------
__global__ void k_normalize(uint16_t* __restrict__ qb, uint16_t* __restrict__ kb) {
  size_t o = (size_t)blockIdx.x * DIM + threadIdx.x;  // wave == one head (32 ch)
  float q = bf2f(qb[o]);
  float s = q * q;
  for (int m = 16; m; m >>= 1) s += __shfl_xor(s, m, 32);
  qb[o] = f2bf(q / fmaxf(sqrtf(s), 1e-12f));
  float k = bf2f(kb[o]);
  float s2 = k * k;
  for (int m = 16; m; m >>= 1) s2 += __shfl_xor(s2, m, 32);
  kb[o] = f2bf(k / fmaxf(sqrtf(s2), 1e-12f));
}

// ---------------- K3: spatial kvs (K=2048) + ksum, write 32x48 ext ---------
__global__ void k_spatial_kvs(const uint16_t* __restrict__ kb,
                              const float* __restrict__ vf,
                              uint16_t* __restrict__ kvs) {
  __shared__ uint16_t ldsA[2 * 512];   // k^T tiles, A-frag order
  __shared__ uint16_t ldsB[2 * 512];   // v tiles, B-frag order
  __shared__ float ksum[32];
  const int tid = threadIdx.x;
  const int h = blockIdx.x & 7, bt = blockIdx.x >> 3;
  if (tid < 32) ksum[tid] = 0.f;
  const int w = tid >> 5, lane = tid & 31, khalf = lane >> 4;
  const int mi = w >> 1, ni = w & 1;
  float ks_acc = 0.f;
  v8f c = {};
  for (int l0 = 0; l0 < NS; l0 += 32) {
    __syncthreads();
    for (int i = 0; i < 8; ++i) {      // stage 32 tok x 32 ch of k and v
      int idx = tid + i * 128;
      int tok = idx >> 5, ch = idx & 31;
      size_t ro = (size_t)(bt * NS + l0 + tok) * DIM + h * HD + ch;
      uint16_t kv = kb[ro];
      ldsA[(ch >> 4) * 512 + (((tok & 8) ? 16 : 0) + (ch & 15)) * 16 +
           (((tok & 16) >> 1) | (tok & 7))] = kv;
      ks_acc += bf2f(kv);
      ldsB[(ch >> 4) * 512 + (((tok & 8) ? 16 : 0) + (ch & 15)) * 16 + (((tok & 16) >> 1) | (tok & 7))] =
          f2bf(vf[ro]);
    }
    __syncthreads();
    Frag a, b;
    a.q[0] = *(const uint4*)&ldsA[mi * 512 + lane * 16];
    a.q[1] = *(const uint4*)&ldsA[mi * 512 + lane * 16 + 8];
    b.q[0] = *(const uint4*)&ldsB[ni * 512 + lane * 16];
    b.q[1] = *(const uint4*)&ldsB[ni * 512 + lane * 16 + 8];
    c = __builtin_amdgcn_wmma_f32_16x16x32_bf16(false, a.v, false, b.v,
                                                (short)0, c, false, false);
  }
  atomicAdd(&ksum[tid & 31], ks_acc);  // each thread staged a fixed channel
  __syncthreads();
  uint16_t* base = kvs + (size_t)(bt * 8 + h) * (32 * 48);
  for (int r = 0; r < 8; ++r)
    base[(mi * 16 + khalf * 8 + r) * 48 + ni * 16 + (lane & 15)] = f2bf(c[r]);
  if (tid < 32) base[tid * 48 + 32] = f2bf(ksum[tid]);
  for (int idx = tid; idx < 32 * 15; idx += 128)
    base[(idx / 15) * 48 + 33 + idx % 15] = 0;
}

// ---------------- K4: spatial num = qhat @ kvs_ext, epilogue ---------------
__global__ void k_spatial_num(const uint16_t* __restrict__ qb,
                              const float* __restrict__ vf,
                              const uint16_t* __restrict__ kvs,
                              uint16_t* __restrict__ ccat) {
  __shared__ uint16_t ldsB[3 * 512];
  const int tid = threadIdx.x;
  const int ngrp = blockIdx.x & 31;
  const int h = (blockIdx.x >> 5) & 7;
  const int bt = blockIdx.x >> 8;
  const uint16_t* src = kvs + (size_t)(bt * 8 + h) * (32 * 48);
  for (int i = 0; i < 12; ++i) {       // stage kvs_ext (32x48), B-frag order
    int idx = tid + i * 128;
    int m = idx / 48, d = idx % 48;
    ldsB[(d >> 4) * 512 + (((m & 8) ? 16 : 0) + (d & 15)) * 16 + (((m & 16) >> 1) | (m & 7))] = src[idx];
  }
  __syncthreads();
  const int w = tid >> 5, lane = tid & 31, khalf = lane >> 4;
  const int tb = bt * NS + ngrp * 64 + w * 16;
  Frag a, b;
  const uint16_t* ap = qb + (size_t)(tb + (lane & 15)) * DIM + h * HD + khalf * 8;
  a.q[0] = *(const uint4*)(ap);
  a.q[1] = *(const uint4*)(ap + 16);
  v8f c0 = {}, c1 = {}, c2 = {};
  b.q[0] = *(const uint4*)&ldsB[lane * 16];
  b.q[1] = *(const uint4*)&ldsB[lane * 16 + 8];
  c0 = __builtin_amdgcn_wmma_f32_16x16x32_bf16(false, a.v, false, b.v, (short)0, c0, false, false);
  b.q[0] = *(const uint4*)&ldsB[512 + lane * 16];
  b.q[1] = *(const uint4*)&ldsB[512 + lane * 16 + 8];
  c1 = __builtin_amdgcn_wmma_f32_16x16x32_bf16(false, a.v, false, b.v, (short)0, c1, false, false);
  b.q[0] = *(const uint4*)&ldsB[1024 + lane * 16];
  b.q[1] = *(const uint4*)&ldsB[1024 + lane * 16 + 8];
  c2 = __builtin_amdgcn_wmma_f32_16x16x32_bf16(false, a.v, false, b.v, (short)0, c2, false, false);
  for (int r = 0; r < 8; ++r) {
    int trow = tb + khalf * 8 + r;
    float den = fmaxf(__shfl(c2[r], lane & 16, 32) + 2048.0f, 1e-5f);
    int ch = h * HD + (lane & 15);
    float v0 = vf[(size_t)trow * DIM + ch];
    ccat[(size_t)trow * FF + ch] = f2bf((c0[r] + 2048.0f * v0) / den);
    float v1 = vf[(size_t)trow * DIM + ch + 16];
    ccat[(size_t)trow * FF + ch + 16] = f2bf((c1[r] + 2048.0f * v1) / den);
  }
}

// ---------------- K5: fused temporal attention (L=24, padded K=32) ---------
__global__ void k_temporal(const uint16_t* __restrict__ qb,
                           const uint16_t* __restrict__ kb,
                           const float* __restrict__ vf,
                           uint16_t* __restrict__ ccat) {
  __shared__ uint16_t lds5[4 * 4608];
  const int tid = threadIdx.x;
  const int lane = tid & 31, wv = tid >> 5, khalf = lane >> 4;
  const int wid = blockIdx.x * 4 + wv;
  const int h = wid & 7;
  const int n = (wid >> 3) & (NS - 1);
  const int b = wid >> 14;
  uint16_t* Q  = lds5 + wv * 4608;  // [32][32] row-major (zero-padded rows)
  uint16_t* AK = Q + 1024;          // 2x512 A-frag order (k^T)
  uint16_t* BV = AK + 1024;         // 2x512 B-frag order (v)
  uint16_t* KE = BV + 1024;         // 3x512 B-frag order (kvs_ext)
  for (int i = lane; i < 4608; i += 32) Q[i] = 0;
  __syncthreads();
  float ksum = 0.f;
  const size_t base = ((size_t)(b * TT) * NS + n) * DIM + h * HD + lane;
  for (int t = 0; t < TT; ++t) {
    size_t ro = base + (size_t)t * NS * DIM;
    Q[t * 32 + lane] = qb[ro];
    uint16_t kv = kb[ro];
    AK[(lane >> 4) * 512 + (((t & 8) ? 16 : 0) + (lane & 15)) * 16 +
       (((t & 16) >> 1) | (t & 7))] = kv;
    ksum += bf2f(kv);
    BV[(lane >> 4) * 512 + (((t & 8) ? 16 : 0) + (lane & 15)) * 16 + (((t & 16) >> 1) | (t & 7))] =
        f2bf(vf[ro]);
  }
  __syncthreads();
  Frag a, bf;
  v8f cc[4];
  for (int mi = 0; mi < 2; ++mi) {
    a.q[0] = *(const uint4*)&AK[mi * 512 + lane * 16];
    a.q[1] = *(const uint4*)&AK[mi * 512 + lane * 16 + 8];
    for (int ni = 0; ni < 2; ++ni) {
      bf.q[0] = *(const uint4*)&BV[ni * 512 + lane * 16];
      bf.q[1] = *(const uint4*)&BV[ni * 512 + lane * 16 + 8];
      v8f z = {};
      cc[mi * 2 + ni] = __builtin_amdgcn_wmma_f32_16x16x32_bf16(
          false, a.v, false, bf.v, (short)0, z, false, false);
    }
  }
  __syncthreads();
  for (int mi = 0; mi < 2; ++mi)
    for (int ni = 0; ni < 2; ++ni)
      for (int r = 0; r < 8; ++r) {
        int m = mi * 16 + khalf * 8 + r;
        int d = ni * 16 + (lane & 15);
        KE[(d >> 4) * 512 + (((m & 8) ? 16 : 0) + (d & 15)) * 16 + (((m & 16) >> 1) | (m & 7))] =
            f2bf(cc[mi * 2 + ni][r]);
      }
  KE[1024 + ((lane & 8) ? 16 : 0) * 16 + (((lane & 16) >> 1) | (lane & 7))] = f2bf(ksum);  // col 32 = ksum
  __syncthreads();
  for (int mi = 0; mi < 2; ++mi) {
    const uint16_t* qrow = Q + (mi * 16 + (lane & 15)) * 32 + khalf * 8;
    a.q[0] = *(const uint4*)(qrow);
    a.q[1] = *(const uint4*)(qrow + 16);
    v8f n0 = {}, n1 = {}, n2 = {};
    bf.q[0] = *(const uint4*)&KE[lane * 16];
    bf.q[1] = *(const uint4*)&KE[lane * 16 + 8];
    n0 = __builtin_amdgcn_wmma_f32_16x16x32_bf16(false, a.v, false, bf.v, (short)0, n0, false, false);
    bf.q[0] = *(const uint4*)&KE[512 + lane * 16];
    bf.q[1] = *(const uint4*)&KE[512 + lane * 16 + 8];
    n1 = __builtin_amdgcn_wmma_f32_16x16x32_bf16(false, a.v, false, bf.v, (short)0, n1, false, false);
    bf.q[0] = *(const uint4*)&KE[1024 + lane * 16];
    bf.q[1] = *(const uint4*)&KE[1024 + lane * 16 + 8];
    n2 = __builtin_amdgcn_wmma_f32_16x16x32_bf16(false, a.v, false, bf.v, (short)0, n2, false, false);
    for (int r = 0; r < 8; ++r) {
      int t = mi * 16 + khalf * 8 + r;
      if (t < TT) {
        size_t row = (size_t)(b * TT + t) * NS + n;
        float den = fmaxf(__shfl(n2[r], lane & 16, 32) + 24.0f, 1e-5f);
        int ch = h * HD + (lane & 15);
        float v0 = vf[row * DIM + ch];
        ccat[row * FF + 256 + ch] = f2bf((n0[r] + 24.0f * v0) / den);
        float v1 = vf[row * DIM + ch + 16];
        ccat[row * FF + 256 + ch + 16] = f2bf((n1[r] + 24.0f * v1) / den);
      }
    }
  }
}

// ---------------- K6: output GEMM (196608x512 @ 512x256) + bias ------------
__global__ void k_out_gemm(const uint16_t* __restrict__ ccat,
                           const uint16_t* __restrict__ woutT,
                           const float* __restrict__ bout,
                           float* __restrict__ out) {
  __shared__ uint16_t ldsB[16 * 512];
  const int tid = threadIdx.x;
  const int n0 = blockIdx.y * 16;
  const int m0 = blockIdx.x * 128;
  for (int i = 0; i < 32; ++i) {
    int idx = tid + i * 256;           // 512x16 panel
    int k = idx >> 4, col = idx & 15, kk = k & 31;
    ldsB[(k >> 5) * 512 + (((kk & 8) ? 16 : 0) + col) * 16 + (((kk & 16) >> 1) | (kk & 7))] =
        woutT[k * DIM + n0 + col];
  }
  __syncthreads();
  const int w = tid >> 5, lane = tid & 31, khalf = lane >> 4;
  const uint16_t* ap = ccat + (size_t)(m0 + w * 16 + (lane & 15)) * FF + khalf * 8;
  v8f c = {};
  for (int ks = 0; ks < 16; ++ks) {
    Frag a, b;
    a.q[0] = *(const uint4*)(ap + ks * 32);
    a.q[1] = *(const uint4*)(ap + ks * 32 + 16);
    b.q[0] = *(const uint4*)&ldsB[ks * 512 + lane * 16];
    b.q[1] = *(const uint4*)&ldsB[ks * 512 + lane * 16 + 8];
    c = __builtin_amdgcn_wmma_f32_16x16x32_bf16(false, a.v, false, b.v,
                                                (short)0, c, false, false);
  }
  const int col = n0 + (lane & 15);
  const float bias = bout[col];
  for (int r = 0; r < 8; ++r) {
    int row = m0 + w * 16 + khalf * 8 + r;
    out[(size_t)row * DIM + col] = c[r] + bias;
  }
}

extern "C" void kernel_launch(void* const* d_in, const int* in_sizes, int n_in,
                              void* d_out, int out_size, void* d_ws, size_t ws_size,
                              hipStream_t stream) {
  (void)in_sizes; (void)n_in; (void)out_size; (void)ws_size;
  const float* x    = (const float*)d_in[0];
  const float* wqkv = (const float*)d_in[1];
  const float* wout = (const float*)d_in[2];
  const float* bout = (const float*)d_in[3];
  float* out = (float*)d_out;
  char* ws = (char*)d_ws;
  uint16_t* qb    = (uint16_t*)(ws);                 //  96 MB, q (bf16, normalized in place)
  uint16_t* kb    = (uint16_t*)(ws + 100663296);     //  96 MB, k
  float*    vf    = (float*)   (ws + 201326592);     // 192 MB, v (f32)
  uint16_t* xb    = (uint16_t*)(ws + 402653184);     //  96 MB, x bf16
  uint16_t* wqkvT = (uint16_t*)(ws + 503316480);     // 384 KB
  uint16_t* woutT = (uint16_t*)(ws + 503709696);     // 256 KB
  uint16_t* kvs   = (uint16_t*)(ws + 503971840);     // 2.25 MB, spatial kvs_ext
  uint16_t* ccat  = (uint16_t*)(ws + 506331136);     // 192 MB, concat(out_s,out_t)

  k_prep_weights<<<dim3(1280), dim3(256), 0, stream>>>(wqkv, wout, wqkvT, woutT);
  k_x_to_bf<<<dim3(49152), dim3(256), 0, stream>>>(x, xb);
  k_qkv_gemm<<<dim3(1536, 48), dim3(256), 0, stream>>>(xb, wqkvT, qb, kb, vf);
  k_normalize<<<dim3(TOKS), dim3(256), 0, stream>>>(qb, kb);
  k_spatial_kvs<<<dim3(768), dim3(128), 0, stream>>>(kb, vf, kvs);
  k_spatial_num<<<dim3(24576), dim3(128), 0, stream>>>(qb, vf, kvs, ccat);
  k_temporal<<<dim3(16384), dim3(128), 0, stream>>>(qb, kb, vf, ccat);
  k_out_gemm<<<dim3(1536, 16), dim3(256), 0, stream>>>(ccat, woutT, bout, out);
}
